// pLayer_1408749273848
// MI455X (gfx1250) — matrix operand order repaired and checked by reading, not compiled, not run on hardware
//
#include <hip/hip_runtime.h>

typedef __attribute__((ext_vector_type(16))) _Float16 v16h;
typedef __attribute__((ext_vector_type(8)))  _Float16 v8h;
typedef __attribute__((ext_vector_type(8)))  float    v8f;

#define GMIN  0.01f
#define GMAX  0.5f
#define PGMIN 1e-4f

constexpr int Bsz = 256;   // batch
constexpr int NIN = 512;   // n_in
constexpr int Tt  = 32;    // timesteps
constexpr int Oo  = 512;   // n_out
constexpr int Mm  = 514;   // n_in + 2
constexpr int KP  = 544;   // Mm padded to 17*32 for WMMA K-steps
constexpr int NZ  = 4;     // noise instances

union U16 { v16h v; v8h h[2]; };

// ---------------- init: zero scalar accumulators --------------------------
__global__ void k_zero(float* scalarAcc) {
  if (threadIdx.x < 2) scalarAcc[threadIdx.x] = 0.0f;
}

// ---------------- theta property + g (per-column) -------------------------
// one thread per output column o
__global__ void k_theta(const float* __restrict__ th, float* __restrict__ absTheta,
                        float* __restrict__ sAbs, float* __restrict__ colmin,
                        float* __restrict__ gcol, _Float16* __restrict__ GeffT) {
  int o = blockIdx.x * blockDim.x + threadIdx.x;       // 0..511
  float mn = 1e30f;
  for (int m = 0; m < Mm; ++m) mn = fminf(mn, fabsf(th[m * Oo + o]));
  colmin[o] = mn;
  float scale = PGMIN / mn;
  float gc = 0.0f;
  for (int m = 0; m < Mm; ++m) {
    float v = th[m * Oo + o];
    float c = fminf(fmaxf(v, -GMAX), GMAX);
    float t = (fabsf(c) < GMIN) ? 0.0f : c;            // property(theta_)
    float s = (t >= 0.0f) ? 1.0f : -1.0f;              // pos - neg
    float at = fabsf(t);
    absTheta[m * Oo + o] = at;
    sAbs[m * Oo + o] = s * at * (1.0f / NZ);           // fold /N_noise
    float g = fabsf(v) * scale;                        // g from RAW theta_
    gc += g;
    GeffT[(size_t)o * KP + m] = (_Float16)(s * g);
  }
  for (int m = Mm; m < KP; ++m) GeffT[(size_t)o * KP + m] = (_Float16)0.0f;
  gcol[o] = gc;
}

// ---------------- gsum_row[m] = sum_o g[m,o] ------------------------------
__global__ void k_gsum(const float* __restrict__ th, const float* __restrict__ colmin,
                       float* __restrict__ gsum_row) {
  int m = blockIdx.x;                                  // 0..513
  __shared__ float red[256];
  float a = 0.0f;
  for (int o = threadIdx.x; o < Oo; o += 256)
    a += fabsf(th[m * Oo + o]) * (PGMIN / colmin[o]);
  red[threadIdx.x] = a; __syncthreads();
  for (int s = 128; s > 0; s >>= 1) {
    if (threadIdx.x < s) red[threadIdx.x] += red[threadIdx.x + s];
    __syncthreads();
  }
  if (threadIdx.x == 0) gsum_row[m] = red[0];
}

// ---------------- noise pass 1: rcp[t,n,o] = 1/(colsum + 1e-10) -----------
__global__ void k_colsum(const float* __restrict__ noise, const float* __restrict__ absTheta,
                         float* __restrict__ rcp) {
  int tn = blockIdx.x;                                 // 0..127
  int o  = threadIdx.x;                                // 0..511
  const float* np = noise + (size_t)tn * Mm * Oo;
  float acc = 0.0f;
  for (int m = 0; m < Mm; ++m) {
    __builtin_prefetch(np + (size_t)(m + 8) * Oo + o, 0, 1);
    acc += absTheta[m * Oo + o] * (0.2f * np[(size_t)m * Oo + o] + 0.9f);
  }
  rcp[tn * Oo + o] = 1.0f / (acc + 1e-10f);
}

// ---------------- noise pass 2: WeffT[t][o][k] (f16, transposed) ----------
__global__ void k_weff(const float* __restrict__ noise, const float* __restrict__ sAbs,
                       const float* __restrict__ rcp, _Float16* __restrict__ WeffT) {
  int t  = blockIdx.y;
  int mc = blockIdx.x / 8;       // 17 chunks of 32 rows (covers KP)
  int oc = blockIdx.x % 8;       // 8 chunks of 64 cols
  int m0 = mc * 32, o0 = oc * 64;
  __shared__ _Float16 tile[32][72];
  int i = threadIdx.x;
  for (int r = 0; r < 8; ++r) {
    int lin = r * 256 + i;
    int mL = lin >> 6, oL = lin & 63;
    int m = m0 + mL, o = o0 + oL;
    float w = 0.0f;
    if (m < Mm) {
      size_t base = ((size_t)t * NZ) * Mm * Oo + (size_t)m * Oo + o;
      float acc = 0.0f;
      for (int n = 0; n < NZ; ++n)
        acc += (0.2f * noise[base + (size_t)n * Mm * Oo] + 0.9f) * rcp[(t * NZ + n) * Oo + o];
      w = sAbs[m * Oo + o] * acc;
    }
    tile[mL][oL] = (_Float16)w;
  }
  __syncthreads();
  for (int r = 0; r < 8; ++r) {
    int lin = r * 256 + i;
    int oL = lin >> 5, mL = lin & 31;
    WeffT[((size_t)t * Oo + (o0 + oL)) * KP + m0 + mL] = tile[mL][oL];
  }
}

// ---------------- x transpose -> xT[t][b][k] (f16) + t1 partial -----------
__global__ void k_xprep(const float* __restrict__ x, const float* __restrict__ gsum_row,
                        _Float16* __restrict__ xT, float* __restrict__ scalarAcc) {
  int b  = blockIdx.x;
  int m0 = blockIdx.y * 128;
  __shared__ float tile[128][33];
  int i = threadIdx.x;
  float tacc = 0.0f;
  for (int r = 0; r < 16; ++r) {
    int lin = r * 256 + i;
    int mL = lin >> 5, t = lin & 31;
    float v = x[((size_t)b * NIN + m0 + mL) * Tt + t];
    tile[mL][t] = v;
    tacc += v * v * gsum_row[m0 + mL];                 // t1 = sum x^2 * gsum_row[m]
  }
  __syncthreads();
  for (int r = 0; r < 16; ++r) {
    int lin = r * 256 + i;
    int t = lin >> 7, mL = lin & 127;
    xT[((size_t)t * Bsz + b) * KP + m0 + mL] = (_Float16)tile[mL][t];
  }
  __syncthreads();
  float* red = &tile[0][0];
  red[i] = tacc; __syncthreads();
  for (int s = 128; s > 0; s >>= 1) {
    if (i < s) red[i] += red[i + s];
    __syncthreads();
  }
  if (i == 0) atomicAdd(scalarAcc, red[0]);
}

// ---------------- bias / pad columns of xT --------------------------------
__global__ void k_xbias(_Float16* __restrict__ xT) {
  int tb = blockIdx.x * blockDim.x + threadIdx.x;      // 0 .. T*B-1
  _Float16* p = xT + (size_t)tb * KP + NIN;
  p[0] = (_Float16)1.0f;                               // a_ext bias column
  for (int j = 1; j < KP - NIN; ++j) p[j] = (_Float16)0.0f;
}

// ---------------- fused dual GEMM (z and cross) via WMMA ------------------
// grid (2, 8, T), block 256 = 8 waves. Wave: one 16-row tile x four 16-col tiles.
__global__ void k_gemm(const _Float16* __restrict__ xT, const _Float16* __restrict__ WeffT,
                       const _Float16* __restrict__ GeffT, const float* __restrict__ gcol,
                       float* __restrict__ zTmp, float* __restrict__ scalarAcc) {
  int t    = blockIdx.z;
  int wid  = threadIdx.x >> 5;
  int lane = threadIdx.x & 31;
  int r0   = (blockIdx.x * 8 + wid) * 16;
  int c0   = blockIdx.y * 64;
  int lh   = lane & 15;
  int aoff = (lane < 16) ? 0 : 16;                     // A: K {0..15}/{16..31}
  int boff = (lane < 16) ? 0 : 16;                     // B: K {0..15}/{16..31}, N = lane%16
  const _Float16* Arow = xT + ((size_t)t * Bsz + r0 + lh) * KP;
  const _Float16* Wt   = WeffT + (size_t)t * Oo * KP;

  v8f c1[4] = {};   // z accumulators
  v8f c2[4] = {};   // cross (A @ Geff) accumulators

  for (int k0 = 0; k0 < KP; k0 += 32) {
    U16 a;
    a.h[0] = *(const v8h*)(Arow + k0 + aoff);
    a.h[1] = *(const v8h*)(Arow + k0 + aoff + 8);
#pragma unroll
    for (int q = 0; q < 4; ++q) {
      int o = c0 + q * 16 + lh;
      const _Float16* wp = Wt    + (size_t)o * KP + k0 + boff;
      const _Float16* gp = GeffT + (size_t)o * KP + k0 + boff;
      U16 bw, bg;
      bw.h[0] = *(const v8h*)(wp);  bw.h[1] = *(const v8h*)(wp + 8);
      bg.h[0] = *(const v8h*)(gp);  bg.h[1] = *(const v8h*)(gp + 8);
      c1[q] = __builtin_amdgcn_wmma_f32_16x16x32_f16(false, a.v, false, bw.v,
                                                     (short)0, c1[q], false, false);
      c2[q] = __builtin_amdgcn_wmma_f32_16x16x32_f16(false, a.v, false, bg.v,
                                                     (short)0, c2[q], false, false);
    }
  }

  // epilogue: write z, accumulate t2 + cross partial
  float pacc = 0.0f;
  int rowAdd = (lane < 16) ? 0 : 8;
#pragma unroll
  for (int q = 0; q < 4; ++q) {
    int col = c0 + q * 16 + lh;
    float gc = gcol[col];
#pragma unroll
    for (int r = 0; r < 8; ++r) {
      float z = c1[q][r];
      int row = r0 + r + rowAdd;
      zTmp[((size_t)t * Bsz + row) * Oo + col] = z;
      pacc += gc * z * z - 2.0f * z * c2[q][r];
    }
  }
  __shared__ float red[256];
  red[threadIdx.x] = pacc; __syncthreads();
  for (int s = 128; s > 0; s >>= 1) {
    if (threadIdx.x < s) red[threadIdx.x] += red[threadIdx.x + s];
    __syncthreads();
  }
  if (threadIdx.x == 0) atomicAdd(scalarAcc + 1, red[0]);
}

// ---------------- zTmp [T,B,O] -> out [B,O,T] -----------------------------
__global__ void k_zout(const float* __restrict__ zTmp, float* __restrict__ out) {
  int b  = blockIdx.x;
  int o0 = blockIdx.y * 64;
  __shared__ float tile[64][33];
  int i = threadIdx.x;
  for (int r = 0; r < 8; ++r) {
    int lin = r * 256 + i;
    int t = lin >> 6, oL = lin & 63;
    tile[oL][t] = zTmp[((size_t)t * Bsz + b) * Oo + o0 + oL];
  }
  __syncthreads();
  for (int r = 0; r < 8; ++r) {
    int lin = r * 256 + i;
    int oL = lin >> 5, t = lin & 31;
    out[((size_t)b * Oo + (o0 + oL)) * Tt + t] = tile[oL][t];
  }
}

// ---------------- final scalar: power -------------------------------------
__global__ void k_final(const float* __restrict__ scalarAcc,
                        const float* __restrict__ gsum_row, float* __restrict__ out) {
  if (threadIdx.x == 0 && blockIdx.x == 0) {
    float t1 = scalarAcc[0] + (float)(Tt * Bsz) * gsum_row[NIN];  // bias row: sx2 = B per t
    out[(size_t)Bsz * Oo * Tt] = (t1 + scalarAcc[1]) / (float)(Bsz * Tt);
  }
}

extern "C" void kernel_launch(void* const* d_in, const int* in_sizes, int n_in,
                              void* d_out, int out_size, void* d_ws, size_t ws_size,
                              hipStream_t stream) {
  (void)in_sizes; (void)n_in; (void)out_size; (void)ws_size;
  const float* x     = (const float*)d_in[0];   // [B, n_in, T]
  const float* th    = (const float*)d_in[1];   // [M, O]
  const float* noise = (const float*)d_in[2];   // [T, N, M, O]
  float* out = (float*)d_out;

  char* ws = (char*)d_ws;
  float*    absTheta  = (float*)(ws + 0);          // 514*512*4      = 1052672
  float*    sAbs      = (float*)(ws + 1052672);    // 514*512*4
  float*    rcp       = (float*)(ws + 2105344);    // 128*512*4      = 262144
  float*    colmin    = (float*)(ws + 2367488);    // 512*4
  float*    gcol      = (float*)(ws + 2369536);    // 512*4
  float*    gsum_row  = (float*)(ws + 2371584);    // 514*4 (pad)
  float*    scalarAcc = (float*)(ws + 2374144);    // 2*4 (pad 256)
  _Float16* GeffT     = (_Float16*)(ws + 2374400); // 512*544*2      = 557056
  _Float16* xT        = (_Float16*)(ws + 2931456); // 32*256*544*2   = 8912896
  _Float16* WeffT     = (_Float16*)(ws + 11844352);// 32*512*544*2   = 17825792
  float*    zTmp      = (float*)(ws + 29670144);   // 32*256*512*4   = 16777216

  k_zero  <<<1, 32, 0, stream>>>(scalarAcc);
  k_theta <<<2, 256, 0, stream>>>(th, absTheta, sAbs, colmin, gcol, GeffT);
  k_gsum  <<<Mm, 256, 0, stream>>>(th, colmin, gsum_row);
  k_colsum<<<Tt * NZ, 512, 0, stream>>>(noise, absTheta, rcp);
  k_weff  <<<dim3(17 * 8, Tt), 256, 0, stream>>>(noise, sAbs, rcp, WeffT);
  k_xprep <<<dim3(Bsz, 4), 256, 0, stream>>>(x, gsum_row, xT, scalarAcc);
  k_xbias <<<(Tt * Bsz) / 256, 256, 0, stream>>>(xT);
  k_gemm  <<<dim3(2, 8, Tt), 256, 0, stream>>>(xT, WeffT, GeffT, gcol, zTmp, scalarAcc);
  k_zout  <<<dim3(Bsz, 8), 256, 0, stream>>>(zTmp, out);
  k_final <<<1, 32, 0, stream>>>(scalarAcc, gsum_row, out);
}
